// Self_Space_Attention_54082228191806
// MI455X (gfx1250) — matrix operand order repaired and checked by reading, not compiled, not run on hardware
//
#include <hip/hip_runtime.h>
#include <hip/hip_bf16.h>

#define NN 4096   // H*W
#define CD 256    // channels
#define DQ 32     // query/key dim
#define NB 4      // batch

typedef __attribute__((ext_vector_type(16))) __bf16 v16bf;
typedef __attribute__((ext_vector_type(8)))  float  v8f;

union BF16x16 { v16bf v; uint4 q[2]; };

// WMMA wrapper: D = A(16x32 bf16) * B(32x16 bf16) + C(16x16 f32)
__device__ __forceinline__ v8f wmma_bf16(v16bf a, v16bf b, v8f c) {
    return __builtin_amdgcn_wmma_f32_16x16x32_bf16(false, a, false, b, (short)0, c, false, false);
}

// A-operand (16x32, 16-bit): lane L holds row (L&15); k-elems e=0..7 -> base+e,
// e=8..15 -> base+16+(e-8), base = (L<16 ? 0 : 8). Two contiguous 16B runs.
__device__ __forceinline__ v16bf load_Aop(const __bf16* rowbase, int kbase) {
    BF16x16 u;
    u.q[0] = *reinterpret_cast<const uint4*>(rowbase + kbase);
    u.q[1] = *reinterpret_cast<const uint4*>(rowbase + kbase + 16);
    return u.v;
}

// B-operand loader, contiguous k order (not AMD's table order; unused): lane L holds col (L&15); k = (L<16?0:16)+e.
__device__ __forceinline__ v16bf load_Bop(const __bf16* p) {
    BF16x16 u;
    u.q[0] = *reinterpret_cast<const uint4*>(p);
    u.q[1] = *reinterpret_cast<const uint4*>(p + 8);
    return u.v;
}

// ---------------- Stage 0: convert weights f32 -> bf16 ----------------
__global__ void cvt_w_kernel(const float* __restrict__ Wq, const float* __restrict__ Wk,
                             const float* __restrict__ Wv,
                             __bf16* oq, __bf16* ok, __bf16* ov) {
    int i = blockIdx.x * blockDim.x + threadIdx.x;
    if (i < DQ * CD) { oq[i] = (__bf16)Wq[i]; ok[i] = (__bf16)Wk[i]; }
    if (i < CD * CD) { ov[i] = (__bf16)Wv[i]; }
}

// ---------------- Stage 1: Q/K/V projections via WMMA ----------------
// Qt[b][n][32], Kt[b][n][32], Vb[b][c][n]  (all bf16, bias folded in)
__global__ __launch_bounds__(128) void proj_kernel(
        const float* __restrict__ x,
        const __bf16* __restrict__ Wq, const __bf16* __restrict__ Wk,
        const __bf16* __restrict__ Wv,
        const float* __restrict__ bq, const float* __restrict__ bk,
        const float* __restrict__ bv,
        __bf16* __restrict__ Qt, __bf16* __restrict__ Kt, __bf16* __restrict__ Vb) {
    const int wave = threadIdx.x >> 5;
    const int lane = threadIdx.x & 31;
    const int b     = blockIdx.x >> 6;              // 64 WGs per batch
    const int ntile = (blockIdx.x & 63) * 4 + wave; // 0..255
    const int n0    = ntile * 16;
    const int nlo   = lane & 15;
    const bool lohalf = lane < 16;
    const int base  = lohalf ? 0 : 8;

    const float* xb = x + (size_t)b * CD * NN;

    v8f accQ[2], accK[2], accV[16];
    const v8f vzero = {0.f,0.f,0.f,0.f,0.f,0.f,0.f,0.f};
#pragma unroll
    for (int t = 0; t < 2; ++t) { accQ[t] = vzero; accK[t] = vzero; }
#pragma unroll
    for (int t = 0; t < 16; ++t) accV[t] = vzero;

    for (int c0 = 0; c0 < CD; c0 += 32) {
        const int kb = c0 + (lohalf ? 0 : 16);
        // prefetch next chunk's x lines (strided rows: one line each)
        if (c0 + 32 < CD) {
#pragma unroll
            for (int e = 0; e < 16; ++e)
                __builtin_prefetch(&xb[(size_t)(kb + 32 + e) * NN + n0 + nlo], 0, 3);
        }
        // B-operand from x (f32 -> bf16): lane col n = n0+nlo, k = channel
        v16bf xop;
#pragma unroll
        for (int e = 0; e < 16; ++e)
            xop[e] = (__bf16)xb[(size_t)(c0 + base + (e & 7) + ((e >> 3) << 4)) * NN + n0 + nlo];

#pragma unroll
        for (int t = 0; t < 2; ++t) {
            v16bf aq = load_Aop(Wq + (size_t)(t * 16 + nlo) * CD + c0, base);
            accQ[t] = wmma_bf16(aq, xop, accQ[t]);
            v16bf ak = load_Aop(Wk + (size_t)(t * 16 + nlo) * CD + c0, base);
            accK[t] = wmma_bf16(ak, xop, accK[t]);
        }
#pragma unroll
        for (int t = 0; t < 16; ++t) {
            v16bf av = load_Aop(Wv + (size_t)(t * 16 + nlo) * CD + c0, base);
            accV[t] = wmma_bf16(av, xop, accV[t]);
        }
    }

    const int n = n0 + nlo;
#pragma unroll
    for (int t = 0; t < 2; ++t)
#pragma unroll
        for (int v = 0; v < 8; ++v) {
            const int d = t * 16 + base + v;
            Qt[((size_t)b * NN + n) * DQ + d] = (__bf16)(accQ[t][v] + bq[d]);
            Kt[((size_t)b * NN + n) * DQ + d] = (__bf16)(accK[t][v] + bk[d]);
        }
#pragma unroll
    for (int t = 0; t < 16; ++t)
#pragma unroll
        for (int v = 0; v < 8; ++v) {
            const int c = t * 16 + base + v;
            Vb[((size_t)b * CD + c) * NN + n] = (__bf16)(accV[t][v] + bv[c]);
        }
}

// ---------------- Stage 2: flash attention + residual ----------------
// Wave owns 16 queries (i-tile) x all 256 channels. Computes S^T tiles so the
// softmax row axis lands on lanes (cheap reduction) and P^T needs only a
// lane-local repack (no swap) to become the B-operand of O^T = V x P^T.
__global__ __launch_bounds__(128) void attn_kernel(
        const float* __restrict__ x,
        const __bf16* __restrict__ Qt, const __bf16* __restrict__ Kt,
        const __bf16* __restrict__ Vb, float* __restrict__ out) {
    const int wave = threadIdx.x >> 5;
    const int lane = threadIdx.x & 31;
    const int b     = blockIdx.x >> 6;
    const int itile = (blockIdx.x & 63) * 4 + wave;
    const int i0    = itile * 16;
    const int nlo   = lane & 15;
    const bool lohalf = lane < 16;
    const int base  = lohalf ? 0 : 8;

    const __bf16* Qb  = Qt + (size_t)b * NN * DQ;
    const __bf16* Kb  = Kt + (size_t)b * NN * DQ;
    const __bf16* Vbb = Vb + (size_t)b * CD * NN;

    // Q as B-operand of S^T = Kstrip x Q^T (held for the whole kernel)
    const v16bf Qop = load_Aop(Qb + (size_t)(i0 + nlo) * DQ, base);

    v8f acc[16];
    const v8f vzero = {0.f,0.f,0.f,0.f,0.f,0.f,0.f,0.f};
#pragma unroll
    for (int t = 0; t < 16; ++t) acc[t] = vzero;

    float m = -1.0e30f;   // running row max (per lane = per query)
    float l = 0.0f;       // running denominator

    // pipelined K A-operands (first chunk)
    v16bf Ka0 = load_Aop(Kb + (size_t)nlo * DQ, base);
    v16bf Ka1 = load_Aop(Kb + (size_t)(16 + nlo) * DQ, base);

    for (int j0 = 0; j0 < NN; j0 += 32) {
        // S^T tiles: rows j, cols i
        v8f s0 = wmma_bf16(Ka0, Qop, vzero);
        v8f s1 = wmma_bf16(Ka1, Qop, vzero);

        // issue next chunk's K loads + this chunk's first V group while the
        // S^T WMMAs and softmax VALU run; prefetch V one chunk ahead.
        if (j0 + 32 < NN) {
            Ka0 = load_Aop(Kb + (size_t)(j0 + 32 + nlo) * DQ, base);
            Ka1 = load_Aop(Kb + (size_t)(j0 + 48 + nlo) * DQ, base);
#pragma unroll
            for (int t = 0; t < 16; ++t)
                __builtin_prefetch(Vbb + (size_t)(t * 16 + nlo) * NN + j0 + 32, 0, 3);
        }
        v16bf Vst[2][4];
#pragma unroll
        for (int u = 0; u < 4; ++u)
            Vst[0][u] = load_Aop(Vbb + (size_t)(u * 16 + nlo) * NN + j0, base);

        // online softmax over the 32 new keys: per-lane 16 values + partner half
        float mx = s0[0];
#pragma unroll
        for (int v = 0; v < 8; ++v) { mx = fmaxf(mx, s0[v]); mx = fmaxf(mx, s1[v]); }
        mx = fmaxf(mx, __shfl_xor(mx, 16, 32));
        const float mnew = fmaxf(m, mx);
        const float corr = __expf(m - mnew);

        float p0[8], p1[8], lsum = 0.f;
#pragma unroll
        for (int v = 0; v < 8; ++v) {
            p0[v] = __expf(s0[v] - mnew);
            p1[v] = __expf(s1[v] - mnew);
            lsum += p0[v] + p1[v];
        }
        l = l * corr + lsum + __shfl_xor(lsum, 16, 32);
        m = mnew;

        // build P^T B-operand (32 j x 16 i): lane-local, no swap
        v16bf pb;
#pragma unroll
        for (int v = 0; v < 8; ++v) {
            pb[v]     = (__bf16)p0[v]; // k = (L<16?0:8)+v
            pb[v + 8] = (__bf16)p1[v]; // k = (L<16?16:24)+v
        }

        // rescale accumulators (overlaps in-flight V group-0 loads)
#pragma unroll
        for (int t = 0; t < 16; ++t) {
#pragma unroll
            for (int v = 0; v < 8; ++v) acc[t][v] *= corr;
        }

        // O^T tiles += V(16c x 32j) * P^T, double-buffered in groups of 4
#pragma unroll
        for (int g = 0; g < 4; ++g) {
            if (g < 3) {
#pragma unroll
                for (int u = 0; u < 4; ++u)
                    Vst[(g + 1) & 1][u] =
                        load_Aop(Vbb + (size_t)(((g + 1) * 4 + u) * 16 + nlo) * NN + j0, base);
            }
#pragma unroll
            for (int u = 0; u < 4; ++u)
                acc[g * 4 + u] = wmma_bf16(Vst[g & 1][u], pb, acc[g * 4 + u]);
        }
    }

    // epilogue: out = x + O / l
    const float inv_l = 1.0f / l;
    const float* xb = x + (size_t)b * CD * NN;
    float* ob = out + (size_t)b * CD * NN;
    const int i = i0 + nlo;
#pragma unroll
    for (int t = 0; t < 16; ++t)
#pragma unroll
        for (int v = 0; v < 8; ++v) {
            const int c = t * 16 + base + v;
            const size_t idx = (size_t)c * NN + i;
            ob[idx] = xb[idx] + acc[t][v] * inv_l;
        }
}

extern "C" void kernel_launch(void* const* d_in, const int* in_sizes, int n_in,
                              void* d_out, int out_size, void* d_ws, size_t ws_size,
                              hipStream_t stream) {
    const float* x  = (const float*)d_in[0];
    const float* Wq = (const float*)d_in[1];
    const float* bq = (const float*)d_in[2];
    const float* Wk = (const float*)d_in[3];
    const float* bk = (const float*)d_in[4];
    const float* Wv = (const float*)d_in[5];
    const float* bv = (const float*)d_in[6];
    float* out = (float*)d_out;

    char* ws = (char*)d_ws;
    __bf16* Wq_bf = (__bf16*)(ws);                 // 32*256*2   = 16384
    __bf16* Wk_bf = (__bf16*)(ws + 16384);         // 32*256*2   = 16384
    __bf16* Wv_bf = (__bf16*)(ws + 32768);         // 256*256*2  = 131072
    __bf16* Qt    = (__bf16*)(ws + 163840);        // 4*4096*32*2 = 1048576
    __bf16* Kt    = (__bf16*)(ws + 1212416);       // 4*4096*32*2 = 1048576
    __bf16* Vb    = (__bf16*)(ws + 2260992);       // 4*256*4096*2 = 8388608
    // total ws usage: 10,649,600 bytes

    cvt_w_kernel<<<256, 256, 0, stream>>>(Wq, Wk, Wv, Wq_bf, Wk_bf, Wv_bf);
    proj_kernel<<<NB * 64, 128, 0, stream>>>(x, Wq_bf, Wk_bf, Wv_bf, bq, bk, bv,
                                             Qt, Kt, Vb);
    attn_kernel<<<NB * 64, 128, 0, stream>>>(x, Qt, Kt, Vb, out);
}
